// ReModel_Mention_37787122270991
// MI455X (gfx1250) — matrix operand order repaired and checked by reading, not compiled, not run on hardware
//
#include <hip/hip_runtime.h>
#include <math.h>

namespace {

constexpr int B_ = 4, L_ = 512, D_ = 768, H_ = 12, M_ = 48, E_ = 32, P_ = 512, R_ = 97;
constexpr int RP_ = 112;          // R padded to 7*16 (zero-padded weight rows)
constexpr int KC_ = D_ * 64;      // 49152: classifier K (= D*BLOCK)
constexpr int MP_ = 64;           // mention count 48 padded to 64 for K%32==0
constexpr float EPS_ = 1e-30f;

typedef _Float16 half8  __attribute__((ext_vector_type(8)));
typedef _Float16 half16 __attribute__((ext_vector_type(16)));
typedef float    f32x8  __attribute__((ext_vector_type(8)));

__device__ __forceinline__ half16 cat16(half8 lo, half8 hi) {
  return __builtin_shufflevector(lo, hi, 0,1,2,3,4,5,6,7,8,9,10,11,12,13,14,15);
}

// A fragment (16x32 f16) from row-major A (lda halfs), tile (m0,k0).
// Per ISA: lanes 0-15 & 16-31 both hold rows M=0..15; g=0 covers K 0-7,16-23; g=1 covers 8-15,24-31.
__device__ __forceinline__ half16 ldA(const _Float16* A, int lda, int m0, int k0, int r, int g) {
  const _Float16* p = A + (size_t)(m0 + r) * lda + (k0 + 8 * g);
  half8 lo = *reinterpret_cast<const half8*>(p);
  half8 hi = *reinterpret_cast<const half8*>(p + 16);
  return cat16(lo, hi);
}

// B fragment (32x16 f16) from BT (N x K row-major, ldb halfs), tile (n0,k0).
// Lane holds column N=r; g=0 covers K 0-7,16-23; g=1 covers 8-15,24-31 (same as A).
__device__ __forceinline__ half16 ldB(const _Float16* BT, int ldb, int n0, int k0, int r, int g) {
  const _Float16* p = BT + (size_t)(n0 + r) * ldb + (k0 + 8 * g);
  half8 lo = *reinterpret_cast<const half8*>(p);
  half8 hi = *reinterpret_cast<const half8*>(p + 16);
  return cat16(lo, hi);
}

__device__ __forceinline__ f32x8 wmma16(half16 a, half16 b, f32x8 c) {
  return __builtin_amdgcn_wmma_f32_16x16x32_f16(false, a, false, b, (short)0, c, false, false);
}

// ---------------- elementwise / conversion kernels ----------------

__global__ void k_cvt_f16(const float* __restrict__ src, _Float16* __restrict__ dst, int n) {
  int i = blockIdx.x * blockDim.x + threadIdx.x;
  if (i < n) dst[i] = (_Float16)src[i];
}

__global__ void k_attsum(const float* __restrict__ att, float* __restrict__ s) {
  int i = blockIdx.x * blockDim.x + threadIdx.x;              // B*L*L
  if (i >= B_ * L_ * L_) return;
  int b = i / (L_ * L_);
  int rem = i - b * (L_ * L_);
  const float* p = att + (size_t)b * H_ * L_ * L_ + rem;
  float acc = 0.f;
  for (int h = 0; h < H_; ++h) acc += p[(size_t)h * L_ * L_];
  s[i] = acc;
}

__global__ void k_attT16(const float* __restrict__ att, _Float16* __restrict__ dst) {
  int i = blockIdx.x * blockDim.x + threadIdx.x;              // B*H*L*L
  if (i >= B_ * H_ * L_ * L_) return;
  int bh = i / (L_ * L_);
  int rem = i - bh * (L_ * L_);
  int m = rem / L_, l = rem - m * L_;
  dst[i] = (_Float16)att[((size_t)bh * L_ + l) * L_ + m];
}

__global__ void k_ctxT16(const float* __restrict__ ctx, _Float16* __restrict__ dst) {
  int i = blockIdx.x * blockDim.x + threadIdx.x;              // B*D*L
  if (i >= B_ * D_ * L_) return;
  int b = i / (D_ * L_);
  int rem = i - b * (D_ * L_);
  int d = rem / L_, l = rem - d * L_;
  dst[i] = (_Float16)ctx[((size_t)b * L_ + l) * D_ + d];
}

__global__ void k_clasw16(const float* __restrict__ w, _Float16* __restrict__ dst) {
  int i = blockIdx.x * blockDim.x + threadIdx.x;              // RP_*KC_
  if (i >= RP_ * KC_) return;
  int r = i / KC_;
  dst[i] = (r < R_) ? (_Float16)w[i] : (_Float16)0.f;
}

// ---------------- small f32 stage kernels ----------------

// mention[b,m,d] = sum_l mm[b,m,l] * ctx[b,l,d]
__global__ void k_mention(const float* __restrict__ mm, const float* __restrict__ ctx,
                          float* __restrict__ men) {
  int i = blockIdx.x * blockDim.x + threadIdx.x;              // B*M*D
  if (i >= B_ * M_ * D_) return;
  int b = i / (M_ * D_);
  int rem = i - b * (M_ * D_);
  int m = rem / D_, d = rem - m * D_;
  const float* pm = mm + ((size_t)b * M_ + m) * L_;
  const float* pc = ctx + (size_t)b * L_ * D_ + d;
  float acc = 0.f;
  for (int l = 0; l < L_; ++l) acc += pm[l] * pc[(size_t)l * D_];
  men[i] = acc;
}

// mentionT16[b,d,mp] = mention[b,m,d] (m<48) else 0
__global__ void k_mentionT16(const float* __restrict__ men, _Float16* __restrict__ dst) {
  int i = blockIdx.x * blockDim.x + threadIdx.x;              // B*D*MP
  if (i >= B_ * D_ * MP_) return;
  int b = i / (D_ * MP_);
  int rem = i - b * (D_ * MP_);
  int d = rem / MP_, m = rem - d * MP_;
  dst[i] = (m < M_) ? (_Float16)men[((size_t)b * M_ + m) * D_ + d] : (_Float16)0.f;
}

// tmp1[b,m,j] = sum_i mm[b,m,i] * attsum[b,i,j]
__global__ void k_tmp1(const float* __restrict__ mm, const float* __restrict__ as,
                       float* __restrict__ t1) {
  int i = blockIdx.x * blockDim.x + threadIdx.x;              // B*M*L
  if (i >= B_ * M_ * L_) return;
  int b = i / (M_ * L_);
  int rem = i - b * (M_ * L_);
  int m = rem / L_, j = rem - m * L_;
  const float* pm = mm + ((size_t)b * M_ + m) * L_;
  const float* ps = as + (size_t)b * L_ * L_ + j;
  float acc = 0.f;
  for (int ii = 0; ii < L_; ++ii) acc += pm[ii] * ps[(size_t)ii * L_];
  t1[i] = acc;
}

// matt[b,m,n] = sum_j tmp1[b,m,j] * mm[b,n,j]
__global__ void k_matt(const float* __restrict__ t1, const float* __restrict__ mm,
                       float* __restrict__ ma) {
  int i = blockIdx.x * blockDim.x + threadIdx.x;              // B*M*M
  if (i >= B_ * M_ * M_) return;
  int b = i / (M_ * M_);
  int rem = i - b * (M_ * M_);
  int m = rem / M_, n = rem - m * M_;
  const float* p1 = t1 + ((size_t)b * M_ + m) * L_;
  const float* p2 = mm + ((size_t)b * M_ + n) * L_;
  float acc = 0.f;
  for (int j = 0; j < L_; ++j) acc += p1[j] * p2[j];
  ma[i] = acc;
}

// eatt[b,e,n] = sum_m em[b,e,m] * matt[b,m,n]
__global__ void k_eatt(const float* __restrict__ em, const float* __restrict__ ma,
                       float* __restrict__ ea) {
  int i = blockIdx.x * blockDim.x + threadIdx.x;              // B*E*M
  if (i >= B_ * E_ * M_) return;
  int b = i / (E_ * M_);
  int rem = i - b * (E_ * M_);
  int e = rem / M_, n = rem - e * M_;
  const float* pe = em + ((size_t)b * E_ + e) * M_;
  const float* pm = ma + (size_t)b * M_ * M_ + n;
  float acc = 0.f;
  for (int m = 0; m < M_; ++m) acc += pe[m] * pm[(size_t)m * M_];
  ea[i] = acc;
}

// per (b,p): normalized h_att/t_att rows (padded to 64, f16)
__global__ void k_hatt(const float* __restrict__ em, const float* __restrict__ ea,
                       const long long* __restrict__ hts,
                       _Float16* __restrict__ hatt16, _Float16* __restrict__ tatt16) {
  int b = blockIdx.x / P_, p = blockIdx.x % P_;
  long long hi = hts[((size_t)b * P_ + p) * 2];
  long long ti = hts[((size_t)b * P_ + p) * 2 + 1];
  float mask = (hi + ti != 0) ? 1.f : 0.f;
  int e = threadIdx.x;                                        // 64 threads
  float ha = 0.f, ta = 0.f;
  if (e < M_) {
    float hm = em[((size_t)b * E_ + (int)hi) * M_ + e] * mask;
    float tm = em[((size_t)b * E_ + (int)ti) * M_ + e] * mask;
    ha = ea[((size_t)b * E_ + (int)hi) * M_ + e] * tm;
    ta = ea[((size_t)b * E_ + (int)ti) * M_ + e] * hm;
  }
  __shared__ float sh[64], st[64];
  sh[e] = ha; st[e] = ta;
  __syncthreads();
  for (int s = 32; s > 0; s >>= 1) {
    if (e < s) { sh[e] += sh[e + s]; st[e] += st[e + s]; }
    __syncthreads();
  }
  float hinv = 1.f / (sh[0] + EPS_);
  float tinv = 1.f / (st[0] + EPS_);
  size_t o = ((size_t)b * P_ + p) * MP_ + e;
  hatt16[o] = (_Float16)(ha * hinv);
  tatt16[o] = (_Float16)(ta * tinv);
}

// em_tok row (b,e): 48-dot per column then row-normalize over L
__global__ void k_emtok(const float* __restrict__ em, const float* __restrict__ mm,
                        _Float16* __restrict__ out) {
  int b = blockIdx.x / E_, e = blockIdx.x % E_;
  int t = threadIdx.x;                                        // 256 threads, 2 cols each
  __shared__ float red[256];
  float v[2];
  const float* pe = em + ((size_t)b * E_ + e) * M_;
  for (int c = 0; c < 2; ++c) {
    int l = t + c * 256;
    const float* pm = mm + (size_t)b * M_ * L_ + l;
    float acc = 0.f;
    for (int m = 0; m < M_; ++m) acc += pe[m] * pm[(size_t)m * L_];
    v[c] = acc;
  }
  red[t] = v[0] + v[1];
  __syncthreads();
  for (int s = 128; s > 0; s >>= 1) {
    if (t < s) red[t] += red[t + s];
    __syncthreads();
  }
  float inv = 1.f / (red[0] + EPS_);
  for (int c = 0; c < 2; ++c)
    out[((size_t)b * E_ + e) * L_ + t + c * 256] = (_Float16)(v[c] * inv);
}

// ctx_att row (b,p): sum over heads of ent[h,hidx,l]*ent[h,tidx,l], normalize, mask, f16
__global__ void k_ctxatt(const float* __restrict__ ent, const long long* __restrict__ hts,
                         _Float16* __restrict__ out) {
  int b = blockIdx.x / P_, p = blockIdx.x % P_;
  long long hi = hts[((size_t)b * P_ + p) * 2];
  long long ti = hts[((size_t)b * P_ + p) * 2 + 1];
  float mask = (hi + ti != 0) ? 1.f : 0.f;
  int t = threadIdx.x;                                        // 256 threads, 2 cols each
  __shared__ float red[256];
  float v[2];
  for (int c = 0; c < 2; ++c) {
    int l = t + c * 256;
    const float* ph = ent + (((size_t)b * H_) * E_ + (int)hi) * L_ + l;
    const float* pt = ent + (((size_t)b * H_) * E_ + (int)ti) * L_ + l;
    float acc = 0.f;
    for (int h = 0; h < H_; ++h)
      acc += ph[(size_t)h * E_ * L_] * pt[(size_t)h * E_ * L_];
    v[c] = acc;
  }
  red[t] = v[0] + v[1];
  __syncthreads();
  for (int s = 128; s > 0; s >>= 1) {
    if (t < s) red[t] += red[t + s];
    __syncthreads();
  }
  float inv = mask / (red[0] + EPS_);
  for (int c = 0; c < 2; ++c)
    out[((size_t)b * P_ + p) * L_ + t + c * 256] = (_Float16)(v[c] * inv);
}

// ---------------- WMMA GEMM kernels ----------------

// ent_att[b,h] (32 x 512) = em_tok[b] (32 x 512) @ att[b,h] (512 x 512), via attT16.
__global__ void k_entatt(const _Float16* __restrict__ emtok16,
                         const _Float16* __restrict__ attT16, float* __restrict__ ent) {
  int wid = (blockIdx.x * blockDim.x + threadIdx.x) >> 5;
  int lane = threadIdx.x & 31, r = lane & 15, g = lane >> 4;
  int nt = wid % (L_ / 16); wid /= (L_ / 16);
  int mt = wid % 2;
  int bh = wid / 2;
  const _Float16* A = emtok16 + (size_t)(bh / H_) * E_ * L_;
  const _Float16* BT = attT16 + (size_t)bh * L_ * L_;
  f32x8 acc = {};
  for (int k0 = 0; k0 < L_; k0 += 32)
    acc = wmma16(ldA(A, L_, mt * 16, k0, r, g), ldB(BT, L_, nt * 16, k0, r, g), acc);
  int col = nt * 16 + r;
#pragma unroll
  for (int rr = 0; rr < 8; ++rr) {
    int row = mt * 16 + rr + 8 * g;
    ent[((size_t)bh * E_ + row) * L_ + col] = acc[rr];
  }
}

// Generic per-batch GEMM, f16 out: C[b] (512 x 768) = A[b] (512 x K) @ B[b] (K x 768) via BT.
// lda == ldb == K. Used for context_info (K=512) and the two h/t projections (K=64).
__global__ void k_gemm_f16out(const _Float16* __restrict__ A, const _Float16* __restrict__ BT,
                              _Float16* __restrict__ C, int K,
                              size_t strideA, size_t strideB) {
  int wid = (blockIdx.x * blockDim.x + threadIdx.x) >> 5;
  int lane = threadIdx.x & 31, r = lane & 15, g = lane >> 4;
  int nt = wid % (D_ / 16); wid /= (D_ / 16);
  int mt = wid % (P_ / 16);
  int b  = wid / (P_ / 16);
  const _Float16* Ab = A + (size_t)b * strideA;
  const _Float16* Bb = BT + (size_t)b * strideB;
  f32x8 acc = {};
  for (int k0 = 0; k0 < K; k0 += 32)
    acc = wmma16(ldA(Ab, K, mt * 16, k0, r, g), ldB(Bb, K, nt * 16, k0, r, g), acc);
  int col = nt * 16 + r;
#pragma unroll
  for (int rr = 0; rr < 8; ++rr) {
    int row = mt * 16 + rr + 8 * g;
    C[((size_t)b * P_ + row) * D_ + col] = (_Float16)acc[rr];
  }
}

// Fused: out = tanh(A1 @ W1^T + b1 + A2 @ W2^T + b2), all (2048 x 768), W given row-major (N x K).
__global__ void k_lintanh(const _Float16* __restrict__ A1, const _Float16* __restrict__ W1,
                          const float* __restrict__ b1,
                          const _Float16* __restrict__ A2, const _Float16* __restrict__ W2,
                          const float* __restrict__ b2, _Float16* __restrict__ out16) {
  int wid = (blockIdx.x * blockDim.x + threadIdx.x) >> 5;
  int lane = threadIdx.x & 31, r = lane & 15, g = lane >> 4;
  int nt = wid % (D_ / 16);
  int mt = wid / (D_ / 16);                                   // 0..127 over B*P rows
  f32x8 acc = {};
  for (int k0 = 0; k0 < D_; k0 += 32)
    acc = wmma16(ldA(A1, D_, mt * 16, k0, r, g), ldB(W1, D_, nt * 16, k0, r, g), acc);
  for (int k0 = 0; k0 < D_; k0 += 32)
    acc = wmma16(ldA(A2, D_, mt * 16, k0, r, g), ldB(W2, D_, nt * 16, k0, r, g), acc);
  int col = nt * 16 + r;
  float bias = b1[col] + b2[col];
#pragma unroll
  for (int rr = 0; rr < 8; ++rr) {
    int row = mt * 16 + rr + 8 * g;
    out16[(size_t)row * D_ + col] = (_Float16)tanhf(acc[rr] + bias);
  }
}

// Classifier: out[p,r] = sum_k feat[p,k]*clasW[r,k] + clasb[r], feat generated on the fly:
// feat[p, n*4096+i*64+j] = h[p,n,i] * t[p,n,j]. K-chunk of 32 has fixed (n,i), contiguous j.
__global__ void k_clas(const _Float16* __restrict__ h16, const _Float16* __restrict__ t16,
                       const _Float16* __restrict__ w16, const float* __restrict__ clasb,
                       float* __restrict__ out) {
  int wid = (blockIdx.x * blockDim.x + threadIdx.x) >> 5;
  int lane = threadIdx.x & 31, r = lane & 15, g = lane >> 4;
  int nt = wid % (RP_ / 16);
  int mt = wid / (RP_ / 16);                                  // 0..127
  int m0 = mt * 16;
  const _Float16* hrow = h16 + (size_t)(m0 + r) * D_;
  const _Float16* trow = t16 + (size_t)(m0 + r) * D_;
  f32x8 acc = {};
  for (int k0 = 0; k0 < KC_; k0 += 32) {
    int n = k0 >> 12;
    int rem = k0 & 4095;
    int i = rem >> 6;
    int j0 = rem & 63;                                        // 0 or 32
    _Float16 hv = hrow[n * 64 + i];
    const _Float16* tp = trow + n * 64 + j0 + 8 * g;
    half8 lo = *reinterpret_cast<const half8*>(tp);
    half8 hi = *reinterpret_cast<const half8*>(tp + 16);
    half16 av = cat16(lo, hi);
#pragma unroll
    for (int q = 0; q < 16; ++q) av[q] = av[q] * hv;
    half16 bv = ldB(w16, KC_, nt * 16, k0, r, g);
    acc = wmma16(av, bv, acc);
  }
  int col = nt * 16 + r;
  if (col < R_) {
    float bb = clasb[col];
#pragma unroll
    for (int rr = 0; rr < 8; ++rr) {
      int row = m0 + rr + 8 * g;
      out[(size_t)row * R_ + col] = acc[rr] + bb;
    }
  }
}

} // namespace

static inline int cdiv(long long a, int b) { return (int)((a + b - 1) / b); }

extern "C" void kernel_launch(void* const* d_in, const int* in_sizes, int n_in,
                              void* d_out, int out_size, void* d_ws, size_t ws_size,
                              hipStream_t stream) {
  (void)in_sizes; (void)n_in; (void)out_size; (void)ws_size;
  const float* context     = (const float*)d_in[0];
  const float* attention   = (const float*)d_in[1];
  const float* mention_map = (const float*)d_in[2];
  const float* entity_map  = (const float*)d_in[3];
  const long long* hts     = (const long long*)d_in[4];
  const float* hW  = (const float*)d_in[5];
  const float* hb  = (const float*)d_in[6];
  const float* tW  = (const float*)d_in[7];
  const float* tb  = (const float*)d_in[8];
  const float* hcW = (const float*)d_in[9];
  const float* hcb = (const float*)d_in[10];
  const float* tcW = (const float*)d_in[11];
  const float* tcb = (const float*)d_in[12];
  const float* clasW = (const float*)d_in[13];
  const float* clasb = (const float*)d_in[14];
  float* out = (float*)d_out;

  char* base = (char*)d_ws;
  size_t off = 0;
  auto alloc = [&](size_t bytes) -> char* {
    char* p = base + off;
    off = (off + bytes + 255) & ~(size_t)255;
    return p;
  };

  _Float16* attT16   = (_Float16*)alloc((size_t)B_ * H_ * L_ * L_ * 2);
  float*    attsum   = (float*)   alloc((size_t)B_ * L_ * L_ * 4);
  float*    mention  = (float*)   alloc((size_t)B_ * M_ * D_ * 4);
  float*    tmp1     = (float*)   alloc((size_t)B_ * M_ * L_ * 4);
  float*    matt     = (float*)   alloc((size_t)B_ * M_ * M_ * 4);
  float*    eatt     = (float*)   alloc((size_t)B_ * E_ * M_ * 4);
  _Float16* hatt16   = (_Float16*)alloc((size_t)B_ * P_ * MP_ * 2);
  _Float16* tatt16   = (_Float16*)alloc((size_t)B_ * P_ * MP_ * 2);
  _Float16* emtok16  = (_Float16*)alloc((size_t)B_ * E_ * L_ * 2);
  float*    entatt   = (float*)   alloc((size_t)B_ * H_ * E_ * L_ * 4);
  _Float16* ctxatt16 = (_Float16*)alloc((size_t)B_ * P_ * L_ * 2);
  _Float16* ctxT16   = (_Float16*)alloc((size_t)B_ * D_ * L_ * 2);
  _Float16* ctxi16   = (_Float16*)alloc((size_t)B_ * P_ * D_ * 2);
  _Float16* menT16   = (_Float16*)alloc((size_t)B_ * D_ * MP_ * 2);
  _Float16* hpre16   = (_Float16*)alloc((size_t)B_ * P_ * D_ * 2);
  _Float16* tpre16   = (_Float16*)alloc((size_t)B_ * P_ * D_ * 2);
  _Float16* hh16     = (_Float16*)alloc((size_t)B_ * P_ * D_ * 2);
  _Float16* tt16     = (_Float16*)alloc((size_t)B_ * P_ * D_ * 2);
  _Float16* hW16     = (_Float16*)alloc((size_t)D_ * D_ * 2);
  _Float16* tW16     = (_Float16*)alloc((size_t)D_ * D_ * 2);
  _Float16* hcW16    = (_Float16*)alloc((size_t)D_ * D_ * 2);
  _Float16* tcW16    = (_Float16*)alloc((size_t)D_ * D_ * 2);
  _Float16* clW16    = (_Float16*)alloc((size_t)RP_ * KC_ * 2);

  const int T = 256;
  // conversions / transposes
  k_attsum    <<<cdiv((long long)B_ * L_ * L_, T), T, 0, stream>>>(attention, attsum);
  k_attT16    <<<cdiv((long long)B_ * H_ * L_ * L_, T), T, 0, stream>>>(attention, attT16);
  k_ctxT16    <<<cdiv((long long)B_ * D_ * L_, T), T, 0, stream>>>(context, ctxT16);
  k_cvt_f16   <<<cdiv(D_ * D_, T), T, 0, stream>>>(hW, hW16, D_ * D_);
  k_cvt_f16   <<<cdiv(D_ * D_, T), T, 0, stream>>>(tW, tW16, D_ * D_);
  k_cvt_f16   <<<cdiv(D_ * D_, T), T, 0, stream>>>(hcW, hcW16, D_ * D_);
  k_cvt_f16   <<<cdiv(D_ * D_, T), T, 0, stream>>>(tcW, tcW16, D_ * D_);
  k_clasw16   <<<cdiv((long long)RP_ * KC_, T), T, 0, stream>>>(clasW, clW16);
  // mention path
  k_mention   <<<cdiv(B_ * M_ * D_, T), T, 0, stream>>>(mention_map, context, mention);
  k_mentionT16<<<cdiv(B_ * D_ * MP_, T), T, 0, stream>>>(mention, menT16);
  k_tmp1      <<<cdiv(B_ * M_ * L_, T), T, 0, stream>>>(mention_map, attsum, tmp1);
  k_matt      <<<cdiv(B_ * M_ * M_, T), T, 0, stream>>>(tmp1, mention_map, matt);
  k_eatt      <<<cdiv(B_ * E_ * M_, T), T, 0, stream>>>(entity_map, matt, eatt);
  k_hatt      <<<B_ * P_, 64, 0, stream>>>(entity_map, eatt, hts, hatt16, tatt16);
  k_emtok     <<<B_ * E_, 256, 0, stream>>>(entity_map, mention_map, emtok16);
  // WMMA GEMMs
  k_entatt<<<(B_ * H_ * 2 * (L_ / 16)) / 4, 128, 0, stream>>>(emtok16, attT16, entatt);
  k_ctxatt<<<B_ * P_, 256, 0, stream>>>(entatt, hts, ctxatt16);
  k_gemm_f16out<<<(B_ * (P_ / 16) * (D_ / 16)) / 4, 128, 0, stream>>>(
      ctxatt16, ctxT16, ctxi16, L_, (size_t)P_ * L_, (size_t)D_ * L_);
  k_gemm_f16out<<<(B_ * (P_ / 16) * (D_ / 16)) / 4, 128, 0, stream>>>(
      tatt16, menT16, hpre16, MP_, (size_t)P_ * MP_, (size_t)D_ * MP_);   // h = t_att @ mention
  k_gemm_f16out<<<(B_ * (P_ / 16) * (D_ / 16)) / 4, 128, 0, stream>>>(
      hatt16, menT16, tpre16, MP_, (size_t)P_ * MP_, (size_t)D_ * MP_);   // t = h_att @ mention
  k_lintanh<<<((B_ * P_ / 16) * (D_ / 16)) / 4, 128, 0, stream>>>(
      hpre16, hW16, hb, ctxi16, hcW16, hcb, hh16);
  k_lintanh<<<((B_ * P_ / 16) * (D_ / 16)) / 4, 128, 0, stream>>>(
      tpre16, tW16, tb, ctxi16, tcW16, tcb, tt16);
  k_clas<<<((B_ * P_ / 16) * (RP_ / 16)) / 4, 128, 0, stream>>>(hh16, tt16, clW16, clasb, out);
}
